// SSE_2946347565774
// MI455X (gfx1250) — matrix operand order repaired and checked by reading, not compiled, not run on hardware
//
#include <hip/hip_runtime.h>
#include <hip/hip_bf16.h>
#include <stdint.h>

typedef __bf16 bf16;
typedef __attribute__((ext_vector_type(16))) __bf16 v16bf;
typedef __attribute__((ext_vector_type(8)))  __bf16 v8bf;
typedef __attribute__((ext_vector_type(8)))  float  v8f;
typedef __attribute__((ext_vector_type(4)))  unsigned int v4u;
typedef __attribute__((ext_vector_type(8)))  int v8i;
typedef __attribute__((ext_vector_type(4)))  int v4i;

#define BB 4
#define NN 1024
#define CC 2048
#define HHID 4096
#define NHEAD 32
#define DHEAD 64

// GEMM block tile
#define BM 128
#define BN 128
#define BK 32

union FragA { v16bf v; v8bf h[2]; };
union FragB { v16bf v; bf16 e[16]; };

__device__ __forceinline__ bf16 f2bf(float f) {
  union { float f; unsigned u; } c; c.f = f;
  unsigned r = c.u + 0x7FFFu + ((c.u >> 16) & 1u);
  unsigned short hv = (unsigned short)(r >> 16);
  bf16 o; __builtin_memcpy(&o, &hv, 2); return o;
}

// async 16-byte global -> LDS copy (per lane), tracked with ASYNCcnt
__device__ __forceinline__ void async_cp16(void* lds_dst, const void* gsrc) {
  uint32_t loff = (uint32_t)(uintptr_t)lds_dst;   // addrspace(3) offset survives truncation
  asm volatile("global_load_async_to_lds_b128 %0, %1, off"
               :: "v"(loff), "v"(gsrc) : "memory");
}
__device__ __forceinline__ void wait_async0() {
  asm volatile("s_wait_asynccnt 0x0" ::: "memory");
}

// ---------------------------------------------------------------- TDM tile load
// Build a 2-D tensor DMA descriptor (ISA ch.8 D# groups 0/1) and issue
// TENSOR_LOAD_TO_LDS: copies a tile1 x tile0 tile of 2-byte elements from a
// row-major tensor (row stride = stride0 elements) into LDS. TENSORcnt-tracked.
__device__ __forceinline__ void tdm_load_tile_2d(const void* gaddr, uint32_t lds_off,
                                                 uint32_t dim0, uint32_t dim1,
                                                 uint32_t tile0, uint32_t tile1,
                                                 uint64_t stride0) {
  const uint64_t ga = (uint64_t)(uintptr_t)gaddr;
  v4u g0;
  g0[0] = 1u;                                                   // count=1, user mode
  g0[1] = lds_off;                                              // lds_addr  [63:32]
  g0[2] = (uint32_t)ga;                                         // global_addr lo
  g0[3] = ((uint32_t)(ga >> 32) & 0x01FFFFFFu) | 0x80000000u;   // addr hi | type=2
  v8i g1;
  g1[0] = (int)(1u << 16);                                      // data_size=1 (2B), mask=0
  g1[1] = (int)((dim0 & 0xFFFFu) << 16);                        // tensor_dim0[15:0]
  g1[2] = (int)(((dim0 >> 16) & 0xFFFFu) | ((dim1 & 0xFFFFu) << 16));
  g1[3] = (int)(((dim1 >> 16) & 0xFFFFu) | ((tile0 & 0xFFFFu) << 16));
  g1[4] = (int)(tile1 & 0xFFFFu);                               // tile_dim1, tile_dim2=0
  g1[5] = (int)(uint32_t)stride0;                               // dim0_stride lo
  g1[6] = (int)((uint32_t)(stride0 >> 32) & 0xFFFFu);           // dim0_stride hi
  g1[7] = 0;
  v4i z4 = {0, 0, 0, 0};
#if __clang_major__ >= 23
  v8i z8 = {0, 0, 0, 0, 0, 0, 0, 0};
  __builtin_amdgcn_tensor_load_to_lds(g0, g1, z4, z4, z8, 0);
#else
  __builtin_amdgcn_tensor_load_to_lds(g0, g1, z4, z4, 0);
#endif
}

// ---------------------------------------------------------------- cvt f32->bf16
__global__ void cvt_f32_bf16(const float* __restrict__ src, bf16* __restrict__ dst,
                             long long n) {
  long long i = (long long)blockIdx.x * blockDim.x + threadIdx.x;
  long long stride = (long long)gridDim.x * blockDim.x;
  for (; i < n; i += stride) dst[i] = f2bf(src[i]);
}

// ---------------------------------------------------------------- GEMM: out[M,N] = A[M,K] * W[N,K]^T (+bias)(+relu)
// Block: 256 thr = 8 waves. Block tile 128x128, k-step 32. Tiles staged by the
// Tensor Data Mover (one descriptor per tile, issued by wave 0), LDS
// double-buffered so the TDM transfer of tile k+1 overlaps the WMMAs on tile k.
// Wave tile: 32 (M) x 64 (N) = 8 WMMAs/k-step.
__global__ __launch_bounds__(256)
void gemm_bf16_wmma(const bf16* __restrict__ A, const bf16* __restrict__ W,
                    const float* __restrict__ bias,
                    bf16* __restrict__ outb, float* __restrict__ outf,
                    int M, int Ncols, int Kdim, int relu) {
  __shared__ bf16 Al[2][BM * BK];   // 8 KB per buffer
  __shared__ bf16 Bl[2][BN * BK];   // 8 KB per buffer

  const int tid   = threadIdx.x;
  const int lane  = tid & 31;
  const int wid   = tid >> 5;
  const int waveM = wid & 3;        // 4 waves along M (32 rows each)
  const int waveN = wid >> 2;       // 2 waves along N (64 cols each)
  const int mb = blockIdx.y * BM;
  const int nb = blockIdx.x * BN;
  const int lhalf = lane & 15;
  const int hi    = lane >> 4;

  v8f acc[2][4] = {};

  // stage one 128x32 A tile + 128x32 B tile into LDS buffer `buf` via TDM.
  auto stage = [&](int buf, int k0) {
    if (wid == 0) {
      tdm_load_tile_2d(A + (size_t)mb * (size_t)Kdim + k0,
                       (uint32_t)(uintptr_t)&Al[buf][0],
                       (uint32_t)Kdim, (uint32_t)M, BK, BM, (uint64_t)Kdim);
      tdm_load_tile_2d(W + (size_t)nb * (size_t)Kdim + k0,
                       (uint32_t)(uintptr_t)&Bl[buf][0],
                       (uint32_t)Kdim, (uint32_t)Ncols, BK, BN, (uint64_t)Kdim);
    }
  };

  stage(0, 0);
  int buf = 0;
  for (int k0 = 0; k0 < Kdim; k0 += BK) {
    if (wid == 0) __builtin_amdgcn_s_wait_tensorcnt((short)0);  // tile `buf` landed
    __syncthreads();                                            // publish to all waves
    if (k0 + BK < Kdim) stage(buf ^ 1, k0 + BK);                // overlap next tile

#pragma unroll
    for (int sm = 0; sm < 2; ++sm) {
      // A fragment (16x32): lanes0-15 K=0..7 & 16..23, lanes16-31 K=8..15 & 24..31
      FragA a;
      const bf16* ap = &Al[buf][(waveM * 32 + sm * 16 + lhalf) * BK + hi * 8];
      a.h[0] = *(const v8bf*)ap;
      a.h[1] = *(const v8bf*)(ap + 16);
#pragma unroll
      for (int t = 0; t < 4; ++t) {
        // B fragment (32x16): lane=col, lanes0-15 K=0..7 & 16..23, lanes16-31 K=8..15 & 24..31
        const bf16* bp = &Bl[buf][(waveN * 64 + t * 16 + lhalf) * BK + hi * 8];
        FragA bq; bq.h[0] = *(const v8bf*)bp; bq.h[1] = *(const v8bf*)(bp + 16);
        v16bf bfr = bq.v;
        acc[sm][t] = __builtin_amdgcn_wmma_f32_16x16x32_bf16(
            false, a.v, false, bfr, (short)0, acc[sm][t], false, false);
      }
    }
    buf ^= 1;
  }

#pragma unroll
  for (int sm = 0; sm < 2; ++sm) {
#pragma unroll
    for (int t = 0; t < 4; ++t) {
      const int col = nb + waveN * 64 + t * 16 + lhalf;
      const float bv = bias ? bias[col] : 0.0f;
#pragma unroll
      for (int j = 0; j < 8; ++j) {
        const int row = mb + waveM * 32 + sm * 16 + j + hi * 8;
        float v = acc[sm][t][j] + bv;
        if (relu) v = fmaxf(v, 0.0f);
        const size_t idx = (size_t)row * (size_t)Ncols + col;
        if (outb) outb[idx] = f2bf(v);
        if (outf) outf[idx] = v;
      }
    }
  }
}

// ---------------------------------------------------------------- flash attention
// One wave handles one (b, h, 16-row q-tile). Block = 128 thr = 4 waves.
__global__ __launch_bounds__(128)
void attention_wmma(const bf16* __restrict__ Qm, const bf16* __restrict__ Km,
                    const bf16* __restrict__ Vm, float* __restrict__ O) {
  __shared__ bf16 smem[4 * (16 * 32 + 32 * 64)];
  const int lane = threadIdx.x & 31;
  const int wid  = threadIdx.x >> 5;
  bf16* Pl = smem + wid * (16 * 32 + 32 * 64);
  bf16* Vt = Pl + 16 * 32;

  const int qt             = blockIdx.x * 4 + wid;
  const int tiles_per_head = NN / 16;
  const int bh    = qt / tiles_per_head;
  const int qtile = qt % tiles_per_head;
  const int b = bh / NHEAD, h = bh % NHEAD;
  const int m0 = qtile * 16;

  const size_t base  = ((size_t)b * NN) * CC + (size_t)h * DHEAD;
  const int    lhalf = lane & 15, hi = lane >> 4;

  // Q fragments: 16 x 64 -> two 16x32 A fragments
  FragA qf[2];
  {
    const bf16* qrow = Qm + base + (size_t)(m0 + lhalf) * CC;
#pragma unroll
    for (int f = 0; f < 2; ++f) {
      const bf16* qp = qrow + f * 32 + hi * 8;
      qf[f].h[0] = *(const v8bf*)qp;
      qf[f].h[1] = *(const v8bf*)(qp + 16);
    }
  }

  v8f oacc[4] = {};
  float mrow[8], lrow[8];
#pragma unroll
  for (int j = 0; j < 8; ++j) { mrow[j] = -3.0e38f; lrow[j] = 0.0f; }
  const float scale = 0.125f;   // 1/sqrt(64)

  for (int kt = 0; kt < NN; kt += 32) {
    // ---- S = (Q*scale) K^T for 32 keys (two 16-key subtiles), K-dim = d = 64
    v8f S[2];
#pragma unroll
    for (int s = 0; s < 2; ++s) {
      v8f sa = {};
      const bf16* krow = Km + base + (size_t)(kt + s * 16 + lhalf) * CC;
#pragma unroll
      for (int f = 0; f < 2; ++f) {
        FragA kq; kq.h[0] = *(const v8bf*)(krow + f * 32 + hi * 8); kq.h[1] = *(const v8bf*)(krow + f * 32 + hi * 8 + 16);
        v16bf bfr = kq.v;
        sa = __builtin_amdgcn_wmma_f32_16x16x32_bf16(
            false, qf[f].v, false, bfr, (short)0, sa, false, false);
      }
      S[s] = sa;
    }

    __syncthreads();   // previous iteration's LDS reads are done

    // ---- stage V tile (32 keys x 64 d) to LDS asynchronously: lane -> key row,
    //      overlapped with the softmax VALU work below.
    {
      const bf16* vrow = Vm + base + (size_t)(kt + lane) * CC;
#pragma unroll
      for (int c = 0; c < 8; ++c)
        async_cp16(Vt + lane * 64 + c * 8, vrow + c * 8);
    }

    // ---- online softmax (row stats uniform within each 16-lane half)
#pragma unroll
    for (int j = 0; j < 8; ++j) {
      float s0 = S[0][j] * scale;
      float s1 = S[1][j] * scale;
      float cmax = fmaxf(s0, s1);
#pragma unroll
      for (int off = 8; off >= 1; off >>= 1)
        cmax = fmaxf(cmax, __shfl_xor(cmax, off, 32));
      float newm = fmaxf(mrow[j], cmax);
      float corr = __expf(mrow[j] - newm);
      float p0 = __expf(s0 - newm);
      float p1 = __expf(s1 - newm);
      float psum = p0 + p1;
#pragma unroll
      for (int off = 8; off >= 1; off >>= 1)
        psum += __shfl_xor(psum, off, 32);
      lrow[j] = lrow[j] * corr + psum;
      mrow[j] = newm;
#pragma unroll
      for (int t = 0; t < 4; ++t) oacc[t][j] *= corr;
      const int r = j + hi * 8;                    // C-layout row
      Pl[r * 32 + lhalf]      = f2bf(p0);          // keys kt+0..15
      Pl[r * 32 + 16 + lhalf] = f2bf(p1);          // keys kt+16..31
    }

    wait_async0();
    __syncthreads();

    // ---- re-read P as a 16x32 A fragment from LDS
    FragA pf;
    {
      const bf16* pp = Pl + lhalf * 32 + hi * 8;
      pf.h[0] = *(const v8bf*)pp;
      pf.h[1] = *(const v8bf*)(pp + 16);
    }

    // ---- AV: 4 WMMAs over d (four 16-col groups); V B-frags gathered from LDS
#pragma unroll
    for (int t = 0; t < 4; ++t) {
      FragB vf;
      const int dcol = t * 16 + lhalf;
#pragma unroll
      for (int e = 0; e < 16; ++e) {
        const int k = (e < 8) ? hi * 8 + e : 16 + hi * 8 + (e - 8);
        vf.e[e] = Vt[k * 64 + dcol];
      }
      oacc[t] = __builtin_amdgcn_wmma_f32_16x16x32_bf16(
          false, pf.v, false, vf.v, (short)0, oacc[t], false, false);
    }
  }

  // ---- normalize and write O (f32, [B*N, C] with head offset)
#pragma unroll
  for (int t = 0; t < 4; ++t) {
    const int col = h * DHEAD + t * 16 + lhalf;
#pragma unroll
    for (int j = 0; j < 8; ++j) {
      const int row = m0 + j + hi * 8;
      O[((size_t)b * NN + row) * CC + col] = oacc[t][j] / lrow[j];
    }
  }
}

// ---------------------------------------------------------------- residual + LayerNorm
// out = LN(x + y) * w + b ; one block per row, C = 2048, 256 threads * 8 elems
__global__ __launch_bounds__(256)
void resid_ln(const float* __restrict__ x, const float* __restrict__ y,
              const float* __restrict__ w, const float* __restrict__ b,
              float* __restrict__ out32, bf16* __restrict__ out16) {
  __shared__ float red[256];
  const int row = blockIdx.x;
  const int tid = threadIdx.x;
  const size_t baseo = (size_t)row * CC;
  float vals[8];
  float s = 0.f;
#pragma unroll
  for (int i = 0; i < 8; ++i) {
    const int c = tid + i * 256;
    const float v = x[baseo + c] + y[baseo + c];
    vals[i] = v; s += v;
  }
  red[tid] = s; __syncthreads();
  for (int o = 128; o > 0; o >>= 1) { if (tid < o) red[tid] += red[tid + o]; __syncthreads(); }
  const float mu = red[0] / (float)CC;
  __syncthreads();
  float sq = 0.f;
#pragma unroll
  for (int i = 0; i < 8; ++i) { const float d = vals[i] - mu; sq += d * d; }
  red[tid] = sq; __syncthreads();
  for (int o = 128; o > 0; o >>= 1) { if (tid < o) red[tid] += red[tid + o]; __syncthreads(); }
  const float rs = rsqrtf(red[0] / (float)CC + 1e-6f);
#pragma unroll
  for (int i = 0; i < 8; ++i) {
    const int c = tid + i * 256;
    const float v = (vals[i] - mu) * rs * w[c] + b[c];
    if (out32) out32[baseo + c] = v;
    if (out16) out16[baseo + c] = f2bf(v);
  }
}

// ---------------------------------------------------------------- launch
extern "C" void kernel_launch(void* const* d_in, const int* in_sizes, int n_in,
                              void* d_out, int out_size, void* d_ws, size_t ws_size,
                              hipStream_t stream) {
  const float* x    = (const float*)d_in[0];
  const float* wq   = (const float*)d_in[1];
  const float* wk   = (const float*)d_in[2];
  const float* wv   = (const float*)d_in[3];
  const float* ln1w = (const float*)d_in[4];
  const float* ln1b = (const float*)d_in[5];
  const float* fc1w = (const float*)d_in[6];
  const float* fc1b = (const float*)d_in[7];
  const float* fc2w = (const float*)d_in[8];
  const float* fc2b = (const float*)d_in[9];
  const float* ln2w = (const float*)d_in[10];
  const float* ln2b = (const float*)d_in[11];
  float* out = (float*)d_out;

  char* ws = (char*)d_ws;
  size_t off = 0;
  auto alloc = [&](size_t bytes) -> void* {
    void* p = ws + off;
    off += (bytes + 255) & ~(size_t)255;
    return p;
  };
  const size_t MN = (size_t)BB * NN;                 // 4096 rows
  bf16* xb    = (bf16*)alloc(MN * CC * 2);
  bf16* wqb   = (bf16*)alloc((size_t)CC * CC * 2);
  bf16* wkb   = (bf16*)alloc((size_t)CC * CC * 2);
  bf16* wvb   = (bf16*)alloc((size_t)CC * CC * 2);
  bf16* fc1wb = (bf16*)alloc((size_t)HHID * CC * 2);
  bf16* fc2wb = (bf16*)alloc((size_t)CC * HHID * 2);
  bf16* Qb    = (bf16*)alloc(MN * CC * 2);
  bf16* Kb    = (bf16*)alloc(MN * CC * 2);
  bf16* Vb    = (bf16*)alloc(MN * CC * 2);
  float* Ob   = (float*)alloc(MN * CC * 4);
  float* y32  = (float*)alloc(MN * CC * 4);
  bf16* yb    = (bf16*)alloc(MN * CC * 2);
  bf16* h1    = (bf16*)alloc(MN * HHID * 2);
  float* h2   = (float*)alloc(MN * CC * 4);
  (void)ws_size; (void)in_sizes; (void)n_in; (void)out_size;

  auto cvt = [&](const float* s, bf16* d, size_t n) {
    cvt_f32_bf16<<<dim3(2048), dim3(256), 0, stream>>>(s, d, (long long)n);
  };
  cvt(x,    xb,    MN * CC);
  cvt(wq,   wqb,   (size_t)CC * CC);
  cvt(wk,   wkb,   (size_t)CC * CC);
  cvt(wv,   wvb,   (size_t)CC * CC);
  cvt(fc1w, fc1wb, (size_t)HHID * CC);
  cvt(fc2w, fc2wb, (size_t)CC * HHID);

  auto gemm = [&](const bf16* A, const bf16* W, const float* bias,
                  bf16* ob, float* of, int M, int Nc, int Kd, int relu) {
    dim3 grid(Nc / BN, M / BM);
    gemm_bf16_wmma<<<grid, dim3(256), 0, stream>>>(A, W, bias, ob, of, M, Nc, Kd, relu);
  };

  // QKV projections
  gemm(xb, wqb, nullptr, Qb, nullptr, (int)MN, CC, CC, 0);
  gemm(xb, wkb, nullptr, Kb, nullptr, (int)MN, CC, CC, 0);
  gemm(xb, wvb, nullptr, Vb, nullptr, (int)MN, CC, CC, 0);

  // attention: B*H*(N/16) q-tiles / 4 waves per block
  attention_wmma<<<dim3((BB * NHEAD * (NN / 16)) / 4), dim3(128), 0, stream>>>(Qb, Kb, Vb, Ob);

  // LN1(x + attn)
  resid_ln<<<dim3((unsigned)MN), dim3(256), 0, stream>>>(x, Ob, ln1w, ln1b, y32, yb);

  // MLP
  gemm(yb, fc1wb, fc1b, h1, nullptr, (int)MN, HHID, CC, 1);
  gemm(h1, fc2wb, fc2b, nullptr, h2, (int)MN, CC, HHID, 0);

  // LN2(y + mlp) -> final output (f32)
  resid_ln<<<dim3((unsigned)MN), dim3(256), 0, stream>>>(y32, h2, ln2w, ln2b, out, (bf16*)nullptr);
}
